// ConstraintModel_71270687310038
// MI455X (gfx1250) — matrix operand order repaired and checked by reading, not compiled, not run on hardware
//
#include <hip/hip_runtime.h>

// ---------------- problem constants ----------------
#define S_LEN 512
#define BATCH 256
#define FEAT  128
#define HID   256          // HC == HG
#define GATES 1024         // 4*HID
#define NTILE 64           // GATES/16 column tiles in proj
#define KC_PAD 160         // 129 padded to multiple of 32
#define KG     384         // FEAT + HID
#define XGSTR  384
#define HSTR   272         // h state LDS stride (bf16), padded
#define GSTR   1032        // gates LDS stride (f32), padded
#define ASTR   144         // mlp hidden LDS stride (bf16), padded

typedef __bf16 bf16;
typedef __attribute__((ext_vector_type(16))) __bf16 frag16;  // WMMA A/B fragment
typedef __attribute__((ext_vector_type(8)))  __bf16 frag8;
typedef __attribute__((ext_vector_type(8)))  float  accf;    // WMMA C/D fragment

__device__ __forceinline__ int lane_id() { return (int)(threadIdx.x & 31); }

// A fragment: 16x32 (MxK) row-major tile. ISA layout: lanes 0-15 hold row M=lane,
// K {0..7,16..23}; lanes 16-31 hold K {8..15,24..31}.
__device__ __forceinline__ frag16 frag_a(const bf16* tile, int ld) {
  int l = lane_id();
  const bf16* p = tile + (size_t)(l & 15) * ld + ((l >> 4) << 3);
  union { frag16 v; frag8 h[2]; } u;
  u.h[0] = *(const frag8*)(p);       // K = kb .. kb+7
  u.h[1] = *(const frag8*)(p + 16);  // K = kb+16 .. kb+23
  return u.v;
}

// B fragment: weight stored as N x K row-major (== B^T). ISA layout: lane holds
// column N=lane&15; lanes 0-15 K {0..7,16..23}, lanes 16-31 K {8..15,24..31}.
__device__ __forceinline__ frag16 frag_b(const bf16* wt, int ld) {
  int l = lane_id();
  const bf16* p = wt + (size_t)(l & 15) * ld + ((l >> 4) << 3);
  union { frag16 v; frag8 h[2]; } u;
  u.h[0] = *(const frag8*)(p);
  u.h[1] = *(const frag8*)(p + 16);
  return u.v;
}

__device__ __forceinline__ accf wmma_bf16(frag16 a, frag16 b, accf c) {
  return __builtin_amdgcn_wmma_f32_16x16x32_bf16(false, a, false, b, (short)0, c,
                                                 false, false);
}

// C/D tile layout: lane = (m>=8?16:0)+n, vgpr r -> row (lane>=16?8:0)+r
__device__ __forceinline__ void store_tile_f32(float* outp, int ld, accf c) {
  int l = lane_id(); int n = l & 15; int mb = (l >> 4) << 3;
#pragma unroll
  for (int r = 0; r < 8; ++r) outp[(size_t)(mb + r) * ld + n] = c[r];
}

__device__ __forceinline__ void store_tile_bf16_relu(bf16* outp, int ld, accf c) {
  int l = lane_id(); int n = l & 15; int mb = (l >> 4) << 3;
#pragma unroll
  for (int r = 0; r < 8; ++r) {
    float v = c[r]; v = v > 0.f ? v : 0.f;
    outp[(size_t)(mb + r) * ld + n] = (bf16)v;
  }
}

// Fragment-major proj tile slot: 32 lanes x 8 bf16, one 16B store per lane.
__device__ __forceinline__ void store_tile_frag(bf16* slot, accf c, float bias) {
  int l = lane_id();
  union { frag8 v; bf16 e[8]; } u;
#pragma unroll
  for (int r = 0; r < 8; ++r) u.e[r] = (bf16)(c[r] + bias);
  *(frag8*)(slot + (size_t)l * 8) = u.v;
}

// Seed accumulator from fragment-major proj slot (LDS or global): one b128/lane.
__device__ __forceinline__ accf load_c_frag(const bf16* slot) {
  int l = lane_id();
  union { frag8 v; bf16 e[8]; } u;
  u.v = *(const frag8*)(slot + (size_t)l * 8);
  accf c;
#pragma unroll
  for (int r = 0; r < 8; ++r) c[r] = (float)u.e[r];
  return c;
}

// ---------------- CDNA5 async global->LDS copy (ASYNCcnt-tracked) ----------------
// Each active lane copies 16B from its global address to its LDS address.
__device__ __forceinline__ void async_b128_to_lds(unsigned lds_byte_off,
                                                  const bf16* gaddr) {
  asm volatile("global_load_async_to_lds_b128 %0, %1, off"
               :: "v"(lds_byte_off), "v"(gaddr) : "memory");
}
__device__ __forceinline__ void wait_async0() {
  asm volatile("s_wait_asynccnt 0x0" ::: "memory");
}

// ---------------- packing kernels ----------------
__global__ void pack_bf16_pad(const float* __restrict__ src, bf16* __restrict__ dst,
                              int N, int K, int KP) {
  size_t i = (size_t)blockIdx.x * blockDim.x + threadIdx.x;
  if (i >= (size_t)N * KP) return;
  int n = (int)(i / KP), k = (int)(i % KP);
  float v = (k < K) ? src[(size_t)n * K + k] : 0.f;
  dst[i] = (bf16)v;
}

// gen-LSTM input cols 0..127 = seq shifted right one step (zeros at t=0)
__global__ void pack_shift(const float* __restrict__ seq, bf16* __restrict__ xg) {
  size_t i = (size_t)blockIdx.x * blockDim.x + threadIdx.x;  // over S*B*F
  if (i >= (size_t)S_LEN * BATCH * FEAT) return;
  int k = (int)(i % FEAT);
  size_t row = i / FEAT;                 // t*B + b
  int t = (int)(row / BATCH);
  float v = (t == 0) ? 0.f : seq[(row - BATCH) * FEAT + k];
  xg[row * XGSTR + k] = (bf16)v;
}

__global__ void vec_add(const float* __restrict__ a, const float* __restrict__ b,
                        float* __restrict__ o, int n) {
  int i = blockIdx.x * blockDim.x + threadIdx.x;
  if (i < n) o[i] = a[i] + b[i];
}

// ---------------- time-parallel input projection GEMM ----------------
// P = X[M][K] @ W[N][K]^T + bias[N], written in WMMA C-fragment-major tile
// layout: P[row_tile][col_tile][lane][8] bf16 (coalesced 16B per lane).
// Block tile 128x128, 8 waves, each wave 64x32 (4x2 16x16 accumulators).
__global__ __launch_bounds__(256) void gemm_xw_bias(
    const bf16* __restrict__ X, int ldx, const bf16* __restrict__ W, int ldw,
    const float* __restrict__ bias, bf16* __restrict__ P, int K) {
  int n0 = blockIdx.x * 128;
  size_t m0 = (size_t)blockIdx.y * 128;
  int w  = (int)(threadIdx.x >> 5);
  int wm = (w >> 2) * 64;
  int wn = (w & 3) * 32;
  accf acc[4][2];
#pragma unroll
  for (int i = 0; i < 4; ++i)
#pragma unroll
    for (int j = 0; j < 2; ++j) {
      accf z = {0.f, 0.f, 0.f, 0.f, 0.f, 0.f, 0.f, 0.f};
      acc[i][j] = z;
    }
  int KT = K >> 5;
  for (int kt = 0; kt < KT; ++kt) {
    if (kt + 1 < KT) {
      // locality 3 -> near-cache (WGP-scope) prefetch of next K tile
      __builtin_prefetch(X + (m0 + wm) * (size_t)ldx + (kt + 1) * 32, 0, 3);
      __builtin_prefetch(W + (size_t)(n0 + wn) * ldw + (kt + 1) * 32, 0, 3);
    }
    frag16 af[4], bw[2];
#pragma unroll
    for (int i = 0; i < 4; ++i)
      af[i] = frag_a(X + (m0 + wm + i * 16) * (size_t)ldx + kt * 32, ldx);
#pragma unroll
    for (int j = 0; j < 2; ++j)
      bw[j] = frag_b(W + (size_t)(n0 + wn + j * 16) * ldw + kt * 32, ldw);
#pragma unroll
    for (int i = 0; i < 4; ++i)
#pragma unroll
      for (int j = 0; j < 2; ++j) acc[i][j] = wmma_bf16(af[i], bw[j], acc[i][j]);
  }
  int nl = lane_id() & 15;
#pragma unroll
  for (int i = 0; i < 4; ++i) {
    size_t rt = (m0 + wm) / 16 + i;                 // row tile index
#pragma unroll
    for (int j = 0; j < 2; ++j) {
      int ct = (n0 + wn) / 16 + j;                  // col tile index
      float bv = bias[n0 + wn + j * 16 + nl];
      store_tile_frag(P + (rt * NTILE + ct) * 256, acc[i][j], bv);
    }
  }
}

// ---------------- sequential LSTM scan ----------------
// One workgroup per 16 batch rows; h (bf16) and c (f32) resident in LDS.
// gates = proj[t] (bias folded, fragment-major) + h @ Whh^T.
// Next step's proj seeds are double-buffered into per-wave LDS via CDNA5
// async global->LDS loads, overlapped with the current step's WMMA/gate math.
// !IS_GEN: reverse time, write hc into gen input matrix (cols 128..383).
// IS_GEN: forward time, fused MLP head (W1/W2 staged in LDS) -> f32 output.
template <bool IS_GEN>
__global__ __launch_bounds__(256) void lstm_scan(
    const bf16* __restrict__ proj, const bf16* __restrict__ Whh,
    bf16* __restrict__ xg, const bf16* __restrict__ w1g,
    const bf16* __restrict__ w2g, const float* __restrict__ b1,
    const float* __restrict__ b2, float* __restrict__ out) {
  extern __shared__ char smem[];
  float* gates = (float*)smem;                                   // 16 x GSTR f32
  float* cst   = (float*)(smem + (size_t)16 * GSTR * 4);         // 16 x HID f32
  bf16*  hbf   = (bf16*)((char*)cst + (size_t)16 * HID * 4);     // 16 x HSTR bf16
  bf16*  w1s   = hbf + 16 * HSTR;                                // [128][HID]
  bf16*  w2s   = w1s + (IS_GEN ? 128 * HID : 0);                 // [128][128]
  bf16*  abf   = w2s + (IS_GEN ? 128 * 128 : 0);                 // 16 x ASTR
  bf16*  pbuf  = abf + (IS_GEN ? 16 * ASTR : 0);                 // 8w x 2buf x 8t x 256

  const int tid = (int)threadIdx.x;
  const int w = tid >> 5;
  const int l = tid & 31;
  const int brow = blockIdx.x * 16;

  for (int i = tid; i < 16 * HSTR; i += 256) hbf[i] = (bf16)0.f;
  for (int i = tid; i < 16 * HID; i += 256) cst[i] = 0.f;
  if constexpr (IS_GEN) {
    for (int i = tid; i < 128 * HID; i += 256) w1s[i] = w1g[i];
    for (int i = tid; i < 128 * 128; i += 256) w2s[i] = w2g[i];
  }
  __syncthreads();

  // per-wave proj staging slots
  auto pslot = [&](int buf, int j) -> bf16* {
    return pbuf + (size_t)(((w << 1) | buf) * 8 + j) * 256;
  };
  auto issue_proj = [&](int t, int buf) {
    const bf16* tiles = proj + (((size_t)t * BATCH + brow) >> 4) * NTILE * 256;
#pragma unroll
    for (int j = 0; j < 8; ++j) {
      const bf16* g = tiles + (size_t)(w * 8 + j) * 256 + l * 8;
      unsigned loff = (unsigned)(size_t)(pslot(buf, j) + l * 8);
      async_b128_to_lds(loff, g);
    }
  };

  issue_proj(IS_GEN ? 0 : (S_LEN - 1), 0);  // prefetch step 0

  for (int it = 0; it < S_LEN; ++it) {
    int t = IS_GEN ? it : (S_LEN - 1 - it);
    int buf = it & 1;

    // seed accumulators from LDS-staged projection (wait for our async batch)
    wait_async0();
    accf acc[8];
#pragma unroll
    for (int j = 0; j < 8; ++j) acc[j] = load_c_frag(pslot(buf, j));

    // kick off next step's proj staging; overlaps with WMMA + gate math below
    if (it + 1 < S_LEN) issue_proj(IS_GEN ? (it + 1) : (S_LEN - 2 - it), buf ^ 1);

    // recurrent GEMM: gates += h @ Whh^T
#pragma unroll
    for (int kt = 0; kt < HID / 32; ++kt) {
      frag16 a = frag_a(hbf + kt * 32, HSTR);
#pragma unroll
      for (int j = 0; j < 8; ++j) {
        frag16 bw = frag_b(Whh + (size_t)(w * 128 + j * 16) * HID + kt * 32, HID);
        acc[j] = wmma_bf16(a, bw, acc[j]);
      }
    }
#pragma unroll
    for (int j = 0; j < 8; ++j) store_tile_f32(gates + w * 128 + j * 16, GSTR, acc[j]);
    __syncthreads();

    // gate nonlinearity: thread handles column tid for all 16 rows
    {
      int cc = tid;
#pragma unroll 4
      for (int m = 0; m < 16; ++m) {
        float gi = gates[m * GSTR + cc];
        float gf = gates[m * GSTR + 256 + cc];
        float gg = gates[m * GSTR + 512 + cc];
        float go = gates[m * GSTR + 768 + cc];
        float si = 1.f / (1.f + __expf(-gi));
        float sf = 1.f / (1.f + __expf(-gf));
        float so = 1.f / (1.f + __expf(-go));
        float c2 = sf * cst[m * HID + cc] + si * tanhf(gg);
        float h2 = so * tanhf(c2);
        cst[m * HID + cc] = c2;
        hbf[m * HSTR + cc] = (bf16)h2;
        if constexpr (!IS_GEN)
          xg[((size_t)t * BATCH + brow + m) * XGSTR + FEAT + cc] = (bf16)h2;
      }
    }
    __syncthreads();

    if constexpr (IS_GEN) {
      // MLP layer 1: relu(h @ W1^T + b1) -> abf (16 x 128); wave w -> N-tile w
      accf a1;
      float bv1 = b1[w * 16 + (l & 15)];
#pragma unroll
      for (int r = 0; r < 8; ++r) a1[r] = bv1;
#pragma unroll
      for (int kt = 0; kt < HID / 32; ++kt) {
        frag16 a = frag_a(hbf + kt * 32, HSTR);
        frag16 bw = frag_b(w1s + (size_t)(w * 16) * HID + kt * 32, HID);
        a1 = wmma_bf16(a, bw, a1);
      }
      store_tile_bf16_relu(abf + w * 16, ASTR, a1);
      __syncthreads();
      // MLP layer 2: a @ W2^T + b2 -> out (16 x 128)
      accf a2;
      float bv2 = b2[w * 16 + (l & 15)];
#pragma unroll
      for (int r = 0; r < 8; ++r) a2[r] = bv2;
#pragma unroll
      for (int kt = 0; kt < 128 / 32; ++kt) {
        frag16 a = frag_a(abf + kt * 32, ASTR);
        frag16 bw = frag_b(w2s + (size_t)(w * 16) * 128 + kt * 32, 128);
        a2 = wmma_bf16(a, bw, a2);
      }
      store_tile_f32(out + ((size_t)t * BATCH + brow) * FEAT + w * 16, FEAT, a2);
    }
  }
}

// ---------------- host orchestration ----------------
extern "C" void kernel_launch(void* const* d_in, const int* in_sizes, int n_in,
                              void* d_out, int out_size, void* d_ws, size_t ws_size,
                              hipStream_t stream) {
  (void)in_sizes; (void)n_in; (void)out_size; (void)ws_size;
  const float* seq   = (const float*)d_in[0];
  const float* seqc  = (const float*)d_in[1];
  const float* Wih_c = (const float*)d_in[2];
  const float* Whh_c = (const float*)d_in[3];
  const float* bih_c = (const float*)d_in[4];
  const float* bhh_c = (const float*)d_in[5];
  const float* Wih_g = (const float*)d_in[6];
  const float* Whh_g = (const float*)d_in[7];
  const float* bih_g = (const float*)d_in[8];
  const float* bhh_g = (const float*)d_in[9];
  const float* W1 = (const float*)d_in[10];
  const float* b1 = (const float*)d_in[11];
  const float* W2 = (const float*)d_in[12];
  const float* b2 = (const float*)d_in[13];
  float* out = (float*)d_out;

  // workspace carve (256B aligned slots)
  size_t off = 0;
  char* base = (char*)d_ws;
  auto carve = [&](size_t bytes) -> void* {
    void* p = base + off;
    off += (bytes + 255) & ~(size_t)255;
    return p;
  };
  bf16* wihc  = (bf16*)carve((size_t)GATES * KC_PAD * 2);
  bf16* whhc  = (bf16*)carve((size_t)GATES * HID * 2);
  bf16* wihg  = (bf16*)carve((size_t)GATES * KG * 2);
  bf16* whhg  = (bf16*)carve((size_t)GATES * HID * 2);
  bf16* w1b   = (bf16*)carve((size_t)128 * HID * 2);
  bf16* w2b   = (bf16*)carve((size_t)128 * 128 * 2);
  float* biasc = (float*)carve((size_t)GATES * 4);
  float* biasg = (float*)carve((size_t)GATES * 4);
  bf16* Xc = (bf16*)carve((size_t)S_LEN * BATCH * KC_PAD * 2);
  bf16* Xg = (bf16*)carve((size_t)S_LEN * BATCH * KG * 2);
  bf16* Pj = (bf16*)carve((size_t)S_LEN * BATCH * GATES * 2);  // reused, frag-major

  dim3 blk(256);
  auto g1 = [](size_t n) { return dim3((unsigned)((n + 255) / 256)); };
  const size_t SB = (size_t)S_LEN * BATCH;

  // pack weights / inputs to bf16 (K padded, zero-filled)
  pack_bf16_pad<<<g1((size_t)GATES * KC_PAD), blk, 0, stream>>>(Wih_c, wihc, GATES, 129, KC_PAD);
  pack_bf16_pad<<<g1((size_t)GATES * HID), blk, 0, stream>>>(Whh_c, whhc, GATES, HID, HID);
  pack_bf16_pad<<<g1((size_t)GATES * KG), blk, 0, stream>>>(Wih_g, wihg, GATES, KG, KG);
  pack_bf16_pad<<<g1((size_t)GATES * HID), blk, 0, stream>>>(Whh_g, whhg, GATES, HID, HID);
  pack_bf16_pad<<<g1((size_t)128 * HID), blk, 0, stream>>>(W1, w1b, 128, HID, HID);
  pack_bf16_pad<<<g1((size_t)128 * 128), blk, 0, stream>>>(W2, w2b, 128, 128, 128);
  vec_add<<<g1(GATES), blk, 0, stream>>>(bih_c, bhh_c, biasc, GATES);
  vec_add<<<g1(GATES), blk, 0, stream>>>(bih_g, bhh_g, biasg, GATES);
  pack_bf16_pad<<<g1(SB * KC_PAD), blk, 0, stream>>>(seqc, Xc, (int)SB, 129, KC_PAD);
  pack_shift<<<g1(SB * FEAT), blk, 0, stream>>>(seq, Xg);

  dim3 ggrid(GATES / 128, (unsigned)(SB / 128));
  const size_t PBUF = (size_t)8 * 2 * 8 * 256 * 2;  // 64KB async proj staging

  // constraint LSTM: full-chip input projection, then reverse scan (writes hc into Xg)
  gemm_xw_bias<<<ggrid, blk, 0, stream>>>(Xc, KC_PAD, wihc, KC_PAD, biasc, Pj, KC_PAD);
  size_t smemC = (size_t)16 * GSTR * 4 + (size_t)16 * HID * 4 + (size_t)16 * HSTR * 2 + PBUF;
  (void)hipFuncSetAttribute((const void*)lstm_scan<false>,
                            hipFuncAttributeMaxDynamicSharedMemorySize, (int)smemC);
  lstm_scan<false><<<dim3(BATCH / 16), blk, smemC, stream>>>(
      Pj, whhc, Xg, nullptr, nullptr, nullptr, nullptr, nullptr);

  // generation LSTM: projection over [shifted seq | hc], forward scan + fused MLP head
  gemm_xw_bias<<<ggrid, blk, 0, stream>>>(Xg, KG, wihg, KG, biasg, Pj, KG);
  size_t smemG = smemC + (size_t)128 * HID * 2 + (size_t)128 * 128 * 2 + (size_t)16 * ASTR * 2;
  (void)hipFuncSetAttribute((const void*)lstm_scan<true>,
                            hipFuncAttributeMaxDynamicSharedMemorySize, (int)smemG);
  lstm_scan<true><<<dim3(BATCH / 16), blk, smemG, stream>>>(
      Pj, whhg, nullptr, w1b, w2b, b1, b2, out);
}
